// CapsNet_1735166788198
// MI455X (gfx1250) — matrix operand order repaired and checked by reading, not compiled, not run on hardware
//
#include <hip/hip_runtime.h>
#include <hip/hip_bf16.h>

typedef __attribute__((ext_vector_type(16))) _Float16 v16h;
typedef __attribute__((ext_vector_type(8)))  _Float16 h8;
typedef __attribute__((ext_vector_type(8)))  float    v8f;

#if defined(__has_builtin)
#if __has_builtin(__builtin_amdgcn_global_load_async_to_lds_b128)
#define USE_ASYNC_COPY 1
#endif
#endif

#ifdef USE_ASYNC_COPY
typedef __attribute__((__vector_size__(4 * sizeof(int)))) int i4;
typedef __attribute__((address_space(1))) i4* g4p;   // global, 16B granules
typedef __attribute__((address_space(3))) i4* l4p;   // LDS, 16B granules

__device__ __forceinline__ void async_wait0()
{
#if defined(__has_builtin)
#if __has_builtin(__builtin_amdgcn_s_wait_asynccnt)
    __builtin_amdgcn_s_wait_asynccnt(0);
#else
    asm volatile("s_wait_asynccnt 0x0" ::: "memory");
#endif
#else
    asm volatile("s_wait_asynccnt 0x0" ::: "memory");
#endif
}
#endif

// ---------------------------------------------------------------------------
// Generic WMMA GEMM: C[M,N] = act( A[M,Kpad] * B[Kpad,N] + bias[N] )
// B pre-packed in tile layout [kt/32][nblk][n in 64][k in 32] f16, so each
// K-step's B tile is a flat 4KB block -> async global->LDS copy.
// Double-buffered LDS, async copy of tile i+1 overlapped with WMMAs of tile i,
// single barrier per iteration.
// AMODE: 0 = dense f16 row-major (lda%16==0, Kpad==Ktrue)
//        1 = conv1 im2col (data f32, 1x28x28, k=9, s=1 -> 20x20, K=81 pad 96)
//        2 = conv2 im2col NHWC (x1 f16 (b,20,20,256), k=9, s=2 -> 6x6,
//            K axis ordered k = (kh*9+kw)*256 + cin, K=20736)
// OMODE: 0 = f16 row-major, 2 = u scatter (b*9216 + n*36 + s) f32,
//        3 = f32 row-major
// ACT:   0 = none, 1 = relu, 2 = sigmoid
// Block: 128 threads (4 waves), tile 64(M) x 64(N), BK = 32.
// ---------------------------------------------------------------------------
template<int AMODE, int OMODE, int ACT>
__global__ void __launch_bounds__(128)
gemm_wmma(const void* __restrict__ Aptr, const _Float16* __restrict__ Bw,
          const float* __restrict__ bias, void* __restrict__ Outp,
          int M, int N, int Kpad, int Ktrue, int lda)
{
    __shared__ alignas(16) _Float16 As[2][64 * 32];   // [m][k] ping-pong
    __shared__ alignas(16) _Float16 Bs[2][64 * 32];   // [n][k] ping-pong

    const int tid  = threadIdx.x;
    const int lane = tid & 31;
    const int wave = tid >> 5;        // 0..3
    const int hf   = lane >> 4;       // 0/1 half-wave
    const int mrow = lane & 15;

    const int mBlock = blockIdx.x * 64;
    const int nBlock = blockIdx.y * 64;
    const int nBlocks = gridDim.y;

    v8f acc[4] = {v8f{}, v8f{}, v8f{}, v8f{}};

    const int arow  = tid >> 1;            // 0..63
    const int acol0 = (tid & 1) * 16;      // 0/16

    // hoist M-side im2col decomposition (K-loop invariant)
    const int am = mBlock + arow;
    int c1_rowbase = 0;           // AMODE 1: data base for (b, oh, ow)
    size_t c2_rowbase = 0;        // AMODE 2: x1 NHWC base for (b, 2oh, 2ow)
    if (AMODE == 1) {
        const int b = am / 400, s = am - b * 400;
        const int oh = s / 20, ow = s - oh * 20;
        c1_rowbase = b * 784 + oh * 28 + ow;
    } else if (AMODE == 2) {
        const int b = am / 36, s = am - b * 36;
        const int oh = s / 6, ow = s - oh * 6;
        c2_rowbase = (((size_t)b * 20 + 2 * oh) * 20 + 2 * ow) * 256;
    }

    const int nt = Kpad >> 5;   // number of K-steps

    // fill tile `it` (K-step) into LDS buffer `buf` (async when available)
    auto fill = [&](int it, int buf) {
        const int kt = it << 5;
        // ---- B tile: flat 4KB block from pre-tiled layout ----
        const size_t tileIdx = (size_t)it * nBlocks + blockIdx.y;
        const _Float16* btile = Bw + tileIdx * 2048 + tid * 16;
#ifdef USE_ASYNC_COPY
        {
            g4p gp = (g4p)(unsigned long long)(uintptr_t)(const void*)btile;
            l4p lp = (l4p)(unsigned)(uintptr_t)(void*)&Bs[buf][tid * 16];
            __builtin_amdgcn_global_load_async_to_lds_b128(gp, lp, 0, 0);
            __builtin_amdgcn_global_load_async_to_lds_b128(gp + 1, lp + 1, 0, 0);
        }
#else
        {
            h8 lo = *(const h8*)btile;
            h8 hi = *(const h8*)(btile + 8);
            *(h8*)&Bs[buf][tid * 16]     = lo;
            *(h8*)&Bs[buf][tid * 16 + 8] = hi;
        }
#endif
        // ---- A tile ----
        _Float16* dst = &As[buf][arow * 32 + acol0];
        if (AMODE == 1) {
            #pragma unroll
            for (int e = 0; e < 16; ++e) {
                const int k = kt + acol0 + e;
                _Float16 val = (_Float16)0.f;
                if (k < 81) {
                    const int kh = k / 9, kw = k - kh * 9;
                    val = (_Float16)(((const float*)Aptr)
                              [c1_rowbase + kh * 28 + kw]);
                }
                dst[e] = val;
            }
        } else {
            const _Float16* ap;
            if (AMODE == 0) {
                ap = (const _Float16*)Aptr + (size_t)am * lda + kt + acol0;
            } else { // AMODE == 2 : NHWC, 16 contiguous channels
                const int kbase = kt + acol0;
                const int p    = kbase >> 8;        // kh*9+kw (0..80)
                const int cin0 = kbase & 255;
                const int kh = p / 9, kw = p - kh * 9;
                ap = (const _Float16*)Aptr + c2_rowbase
                     + ((size_t)kh * 20 + kw) * 256 + cin0;
            }
#ifdef USE_ASYNC_COPY
            g4p gp = (g4p)(unsigned long long)(uintptr_t)(const void*)ap;
            l4p lp = (l4p)(unsigned)(uintptr_t)(void*)dst;
            __builtin_amdgcn_global_load_async_to_lds_b128(gp, lp, 0, 0);
            __builtin_amdgcn_global_load_async_to_lds_b128(gp + 1, lp + 1, 0, 0);
#else
            *(h8*)dst       = *(const h8*)ap;
            *(h8*)(dst + 8) = *(const h8*)(ap + 8);
#endif
        }
    };

    fill(0, 0);   // prologue: tile 0 in flight

    for (int i = 0; i < nt; ++i) {
        const int cur = i & 1;
#ifdef USE_ASYNC_COPY
        async_wait0();            // my async copies for tile i are in LDS
#endif
        __syncthreads();          // everyone's tile i ready; prev reads done
        if (i + 1 < nt)
            fill(i + 1, cur ^ 1); // overlap next copy with this tile's WMMAs

        // -------- A fragment (16x32 f16, wave32 layout) --------
        const _Float16* ap = &As[cur][(wave * 16 + mrow) * 32];
        h8 alo = *(const h8*)(ap + hf * 8);
        h8 ahi = *(const h8*)(ap + 16 + hf * 8);
        v16h af = __builtin_shufflevector(alo, ahi, 0,1,2,3,4,5,6,7,
                                                    8,9,10,11,12,13,14,15);
        // -------- 4 N-subtiles --------
        #pragma unroll
        for (int j = 0; j < 4; ++j) {
            const _Float16* bp = &Bs[cur][(j * 16 + mrow) * 32 + hf * 8];
            h8 blo = *(const h8*)bp;
            h8 bhi = *(const h8*)(bp + 16);
            v16h bf = __builtin_shufflevector(blo, bhi, 0,1,2,3,4,5,6,7,
                                                        8,9,10,11,12,13,14,15);
            acc[j] = __builtin_amdgcn_wmma_f32_16x16x32_f16(
                         false, af, false, bf, (short)0, acc[j], false, false);
        }
    }

    // -------- epilogue: bias + activation + store --------
    #pragma unroll
    for (int j = 0; j < 4; ++j) {
        const int n = nBlock + j * 16 + mrow;
        if (n >= N) continue;
        const float bv = bias[n];
        #pragma unroll
        for (int e = 0; e < 8; ++e) {
            const int m = mBlock + wave * 16 + e + 8 * hf;
            float val = acc[j][e] + bv;
            if (ACT == 1) val = fmaxf(val, 0.f);
            else if (ACT == 2) val = 1.0f / (1.0f + __expf(-val));
            if (OMODE == 0) {
                ((_Float16*)Outp)[(size_t)m * N + n] = (_Float16)val;
            } else if (OMODE == 2) {
                const int b = m / 36, s = m - b * 36;
                ((float*)Outp)[(size_t)b * 9216 + (size_t)n * 36 + s] = val;
            } else {
                ((float*)Outp)[(size_t)m * N + n] = val;
            }
        }
    }
}

// ---------------------------------------------------------------------------
// Pack f32 weights [N][Ksrc] (row-major) into f16 tile layout
// [kt][nblk][n in 64][k in 32]; zero pad k >= Ktrue or n >= N.
// KMAP 0: ksrc = k.  KMAP 1 (conv2): k = (kh*9+kw)*256+cin -> ksrc = cin*81+p
// ---------------------------------------------------------------------------
template<int KMAP>
__global__ void pack_tiled(const float* __restrict__ src, _Float16* __restrict__ dst,
                           int N, int Ksrc, int Ktrue, int nBlocks, int total)
{
    const int i = blockIdx.x * blockDim.x + threadIdx.x;
    if (i >= total) return;
    const int tile  = i >> 11;
    const int inner = i & 2047;
    const int n   = inner >> 5;
    const int kin = inner & 31;
    const int nb    = tile % nBlocks;
    const int ktile = tile / nBlocks;
    const int k  = ktile * 32 + kin;
    const int ng = nb * 64 + n;
    _Float16 val = (_Float16)0.f;
    if (k < Ktrue && ng < N) {
        const int ks = (KMAP == 1) ? ((k & 255) * 81 + (k >> 8)) : k;
        val = (_Float16)src[(size_t)ng * Ksrc + ks];
    }
    dst[i] = val;
}

__global__ void zero_kern(float* p, int n)
{
    const int i = blockIdx.x * blockDim.x + threadIdx.x;
    if (i < n) p[i] = 0.f;
}

// squash over last axis of u (B*1152, 8)
__global__ void squash_kern(float* __restrict__ u, int rows)
{
    const int i = blockIdx.x * blockDim.x + threadIdx.x;
    if (i >= rows) return;
    float* up = u + (size_t)i * 8;
    float sn = 0.f;
    #pragma unroll
    for (int k = 0; k < 8; ++k) sn += up[k] * up[k];
    const float f = sn / ((1.0f + sn) * sqrtf(sn));
    #pragma unroll
    for (int k = 0; k < 8; ++k) up[k] *= f;
}

// u_hat[b,r,j,o] = sum_i W[r,j,o,i] * u[b,r,i]   (block per (b,r), 160 thr)
__global__ void uhat_kern(const float* __restrict__ u, const float* __restrict__ W,
                          float* __restrict__ uhat)
{
    const int br = blockIdx.x;            // b*1152 + r
    const int r  = br % 1152;
    const int t  = threadIdx.x;           // j*16 + o
    __shared__ float uld[8];
    if (t < 8) uld[t] = u[(size_t)br * 8 + t];
    __syncthreads();
    const float* wp = W + (size_t)r * 1280 + (size_t)t * 8;
    float acc = 0.f;
    #pragma unroll
    for (int i = 0; i < 8; ++i) acc += wp[i] * uld[i];
    uhat[(size_t)br * 160 + t] = acc;
}

// softmax over r (1152) per column j; block per j
__global__ void route_softmax(const float* __restrict__ bij, float* __restrict__ c)
{
    const int j = blockIdx.x;
    const int t = threadIdx.x;
    __shared__ float red[256];
    float m = -1e30f;
    for (int r = t; r < 1152; r += 256) m = fmaxf(m, bij[r * 10 + j]);
    red[t] = m; __syncthreads();
    for (int s = 128; s > 0; s >>= 1) {
        if (t < s) red[t] = fmaxf(red[t], red[t + s]);
        __syncthreads();
    }
    const float mx = red[0]; __syncthreads();
    float sum = 0.f;
    for (int r = t; r < 1152; r += 256) sum += __expf(bij[r * 10 + j] - mx);
    red[t] = sum; __syncthreads();
    for (int s = 128; s > 0; s >>= 1) {
        if (t < s) red[t] += red[t + s];
        __syncthreads();
    }
    const float inv = 1.0f / red[0];
    for (int r = t; r < 1152; r += 256)
        c[r * 10 + j] = __expf(bij[r * 10 + j] - mx) * inv;
}

// s_j then v_j: block per b, 160 threads (t = j*16+o)
__global__ void route_sv(const float* __restrict__ c, const float* __restrict__ uhat,
                         float* __restrict__ v)
{
    const int b = blockIdx.x;
    const int t = threadIdx.x;
    const int j = t >> 4;
    const float* up = uhat + (size_t)b * 184320;
    float acc = 0.f;
    for (int r = 0; r < 1152; ++r)
        acc += c[r * 10 + j] * up[(size_t)r * 160 + t];
    v[(size_t)b * 160 + t] = acc * fabsf(acc) / (1.0f + acc * acc);
}

// b_ij[r,j] += mean_b sum_o u_hat[b,r,j,o]*v[b,j,o]; block per r, 160 threads
__global__ void route_bupd(const float* __restrict__ uhat, const float* __restrict__ v,
                           float* __restrict__ bij)
{
    const int r = blockIdx.x;
    const int t = threadIdx.x;
    __shared__ float red[160];
    float acc = 0.f;
    for (int b = 0; b < 256; ++b)
        acc += uhat[(size_t)b * 184320 + (size_t)r * 160 + t] * v[(size_t)b * 160 + t];
    red[t] = acc; __syncthreads();
    if (t < 10) {
        float s = 0.f;
        #pragma unroll
        for (int o = 0; o < 16; ++o) s += red[t * 16 + o];
        bij[r * 10 + t] += s * (1.0f / 256.0f);
    }
}

// classes[b,j] = sqrt(sum_o v[b,j,o]^2)
__global__ void cls_kern(const float* __restrict__ v, float* __restrict__ cls)
{
    const int i = blockIdx.x * blockDim.x + threadIdx.x;
    if (i >= 2560) return;
    const float* vp = v + (size_t)i * 16;
    float s = 0.f;
    #pragma unroll
    for (int o = 0; o < 16; ++o) s += vp[o] * vp[o];
    cls[i] = sqrtf(s);
}

// per-class (batch axis) softmax stats: block per j, 256 threads = B
__global__ void colstats_kern(const float* __restrict__ cls,
                              float* __restrict__ cmax, float* __restrict__ csum)
{
    const int j = blockIdx.x;
    const int t = threadIdx.x;
    __shared__ float red[256];
    const float x = cls[t * 10 + j];
    red[t] = x; __syncthreads();
    for (int s = 128; s > 0; s >>= 1) {
        if (t < s) red[t] = fmaxf(red[t], red[t + s]);
        __syncthreads();
    }
    const float mx = red[0]; __syncthreads();
    red[t] = __expf(x - mx); __syncthreads();
    for (int s = 128; s > 0; s >>= 1) {
        if (t < s) red[t] += red[t + s];
        __syncthreads();
    }
    if (t == 0) { cmax[j] = mx; csum[j] = red[0]; }
}

// argmax over batch-softmaxed class lengths, one-hot mask, masked caps -> f16
__global__ void mask_kern(const float* __restrict__ v, const float* __restrict__ cls,
                          const float* __restrict__ cmax, const float* __restrict__ csum,
                          float* __restrict__ out_output, float* __restrict__ out_masked,
                          _Float16* __restrict__ tbuf)
{
    const int b = threadIdx.x;
    int best = 0; float bp = -1e30f;
    #pragma unroll
    for (int j = 0; j < 10; ++j) {
        const float p = __expf(cls[b * 10 + j] - cmax[j]) / csum[j];
        if (p > bp) { bp = p; best = j; }
    }
    #pragma unroll
    for (int j = 0; j < 10; ++j)
        out_masked[b * 10 + j] = (j == best) ? 1.0f : 0.0f;
    for (int t = 0; t < 160; ++t) {
        const float val = v[(size_t)b * 160 + t];
        out_output[(size_t)b * 160 + t] = val;
        const int j = t >> 4;
        tbuf[(size_t)b * 160 + t] = (_Float16)((j == best) ? val : 0.0f);
    }
}

// ---------------------------------------------------------------------------
extern "C" void kernel_launch(void* const* d_in, const int* in_sizes, int n_in,
                              void* d_out, int out_size, void* d_ws, size_t ws_size,
                              hipStream_t stream)
{
    const float* data   = (const float*)d_in[0];   // (256,1,28,28)
    const float* conv_w = (const float*)d_in[1];   // (256,1,9,9)
    const float* conv_b = (const float*)d_in[2];   // (256,)
    const float* pc_w   = (const float*)d_in[3];   // (8,32,256,9,9) -> (256,20736)
    const float* pc_b   = (const float*)d_in[4];   // (8,32) -> (256,)
    const float* Wcap   = (const float*)d_in[5];   // (1,1152,10,16,8)
    const float* dec_w1 = (const float*)d_in[6];   // (512,160)
    const float* dec_b1 = (const float*)d_in[7];
    const float* dec_w2 = (const float*)d_in[8];   // (1024,512)
    const float* dec_b2 = (const float*)d_in[9];
    const float* dec_w3 = (const float*)d_in[10];  // (784,1024)
    const float* dec_b3 = (const float*)d_in[11];

    float* out_f      = (float*)d_out;
    float* out_output = out_f;               // 256*10*16 = 40960
    float* out_recon  = out_f + 40960;       // 256*784   = 200704
    float* out_masked = out_f + 241664;      // 256*10    = 2560

    // bump allocator over d_ws
    char* wsp = (char*)d_ws;
    auto alloc = [&](size_t bytes) -> void* {
        void* p = (void*)wsp;
        wsp += (bytes + 255) & ~(size_t)255;
        return p;
    };
    _Float16* w1p  = (_Float16*)alloc((size_t)24576   * 2);  // 3 kt * 4 nb
    _Float16* w2p  = (_Float16*)alloc((size_t)5308416 * 2);  // 648 kt * 4 nb
    _Float16* d1p  = (_Float16*)alloc((size_t)81920   * 2);  // 5 kt * 8 nb
    _Float16* d2p  = (_Float16*)alloc((size_t)524288  * 2);  // 16 kt * 16 nb
    _Float16* d3p  = (_Float16*)alloc((size_t)851968  * 2);  // 32 kt * 13 nb
    _Float16* x1   = (_Float16*)alloc((size_t)256 * 400 * 256 * 2); // NHWC
    float*    u    = (float*)alloc((size_t)256 * 9216 * 4);
    float*    uhat = (float*)alloc((size_t)256 * 184320 * 4);
    float*    bij  = (float*)alloc((size_t)11520 * 4);
    float*    cbuf = (float*)alloc((size_t)11520 * 4);
    float*    vbuf = (float*)alloc((size_t)256 * 160 * 4);
    float*    clsb = (float*)alloc((size_t)2560 * 4);
    float*    cmax = (float*)alloc((size_t)16 * 4);
    float*    csum = (float*)alloc((size_t)16 * 4);
    _Float16* tbuf = (_Float16*)alloc((size_t)256 * 160 * 2);
    _Float16* h1   = (_Float16*)alloc((size_t)256 * 512 * 2);
    _Float16* h2   = (_Float16*)alloc((size_t)256 * 1024 * 2);

    // ---- pack weights into f16 tile layout ----
    pack_tiled<0><<<(24576 + 255) / 256, 256, 0, stream>>>(conv_w, w1p, 256, 81, 81, 4, 24576);
    pack_tiled<1><<<(5308416 + 255) / 256, 256, 0, stream>>>(pc_w, w2p, 256, 20736, 20736, 4, 5308416);
    pack_tiled<0><<<(81920 + 255) / 256, 256, 0, stream>>>(dec_w1, d1p, 512, 160, 160, 8, 81920);
    pack_tiled<0><<<(524288 + 255) / 256, 256, 0, stream>>>(dec_w2, d2p, 1024, 512, 512, 16, 524288);
    pack_tiled<0><<<(851968 + 255) / 256, 256, 0, stream>>>(dec_w3, d3p, 784, 1024, 1024, 13, 851968);
    zero_kern<<<(11520 + 255) / 256, 256, 0, stream>>>(bij, 11520);

    // ---- conv1: M=102400, N=256, K=81 (pad 96), relu, f16 NHWC out ----
    gemm_wmma<1, 0, 1><<<dim3(1600, 4), 128, 0, stream>>>(
        (const void*)data, w1p, conv_b, (void*)x1, 102400, 256, 96, 81, 0);

    // ---- conv2 (primary caps): M=9216, N=256, K=20736, f32 scatter to u ----
    gemm_wmma<2, 2, 0><<<dim3(144, 4), 128, 0, stream>>>(
        (const void*)x1, w2p, pc_b, (void*)u, 9216, 256, 20736, 20736, 0);

    // ---- squash u (B,1152,8) ----
    squash_kern<<<(256 * 1152 + 255) / 256, 256, 0, stream>>>(u, 256 * 1152);

    // ---- u_hat (B,1152,10,16) ----
    uhat_kern<<<256 * 1152, 160, 0, stream>>>(u, Wcap, uhat);

    // ---- dynamic routing (3 iterations) ----
    for (int it = 0; it < 3; ++it) {
        route_softmax<<<10, 256, 0, stream>>>(bij, cbuf);
        route_sv<<<256, 160, 0, stream>>>(cbuf, uhat, vbuf);
        if (it < 2)
            route_bupd<<<1152, 160, 0, stream>>>(uhat, vbuf, bij);
    }

    // ---- classes / batch softmax / argmax / mask ----
    cls_kern<<<10, 256, 0, stream>>>(vbuf, clsb);
    colstats_kern<<<10, 256, 0, stream>>>(clsb, cmax, csum);
    mask_kern<<<1, 256, 0, stream>>>(vbuf, clsb, cmax, csum,
                                     out_output, out_masked, tbuf);

    // ---- decoder: 160 -> 512 -> 1024 -> 784 ----
    gemm_wmma<0, 0, 1><<<dim3(4, 8), 128, 0, stream>>>(
        (const void*)tbuf, d1p, dec_b1, (void*)h1, 256, 512, 160, 160, 160);
    gemm_wmma<0, 0, 1><<<dim3(4, 16), 128, 0, stream>>>(
        (const void*)h1, d2p, dec_b2, (void*)h2, 256, 1024, 512, 512, 512);
    gemm_wmma<0, 3, 2><<<dim3(4, 13), 128, 0, stream>>>(
        (const void*)h2, d3p, dec_b3, (void*)out_recon, 256, 784, 1024, 1024, 1024);
}
